// AttentionLSTM_14886356648088
// MI455X (gfx1250) — matrix operand order repaired and checked by reading, not compiled, not run on hardware
//
#include <hip/hip_runtime.h>
#include <hip/hip_bf16.h>

// ---------------------------------------------------------------------------
// AttentionLSTM on MI455X (gfx1250, wave32, WMMA).
//
// Pipeline (all on `stream`, stream order = dependency order):
//   1. cvt f32->f16 of inputs + weights
//   2. gx = X @ Wih^T + (bih+bhh)      -- WMMA GEMM, 1x4 tile blocking
//   3. LSTM recurrence (persistent WG, WMMA for h @ Whh^T, c in registers)
//      ... x4 (enc L0, enc L1, dec L0, dec L1; dec init from enc final state)
//   4. enc_out transpose (S,B,H)->(B,H,S) via async-to-LDS staging
//   5. scores = dec_out . enc_out (WMMA) -> softmax
//   6. attn_v = attn_w @ enc_out^T (WMMA, contiguous B fragments)
//
// Inner loops are written loads-first / WMMAs-second so the scheduler can
// retire v_wmma against partial s_wait_loadcnt values instead of full drains.
// ---------------------------------------------------------------------------

#define S_LEN 512
#define T_LEN 512
#define BATCH 64
#define INDIM 256
#define HID   256
#define G4H   1024            // 4*HID
#define SB    (S_LEN * BATCH) // 32768 rows for the batched projections

typedef __attribute__((ext_vector_type(16))) _Float16 v16h;
typedef __attribute__((ext_vector_type(8)))  _Float16 v8h;
typedef __attribute__((ext_vector_type(8)))  float    v8f;

// ---- WMMA fragment loaders (layouts per CDNA5 ISA 7.12.2, wave32) ----------

// A operand, 16x32 f16, row-major source. Row = base + (lane&15)*rowStride.
// lane<16 holds K {kb+0..7, kb+16..23}; lane>=16 holds K {kb+8..15, kb+24..31}.
__device__ __forceinline__ v16h load_a16(const _Float16* base, long rowStride,
                                         int kb, int lane) {
    const _Float16* p = base + (long)(lane & 15) * rowStride + kb + ((lane >> 4) << 3);
    v8h lo = *(const v8h*)p;
    v8h hi = *(const v8h*)(p + 16);
    return __builtin_shufflevector(lo, hi, 0, 1, 2, 3, 4, 5, 6, 7,
                                           8, 9, 10, 11, 12, 13, 14, 15);
}

// B operand, 32x16 f16, element (k, n) at base + n*colStride + k.
// lane<16 holds K {kb+0..7, kb+16..23}; lane>=16 holds K {kb+8..15, kb+24..31}.
__device__ __forceinline__ v16h load_b16(const _Float16* base, long colStride,
                                         int kb, int lane) {
    const _Float16* p = base + (long)(lane & 15) * colStride + kb + ((lane >> 4) << 3);
    v8h lo = *(const v8h*)p;
    v8h hi = *(const v8h*)(p + 16);
    return __builtin_shufflevector(lo, hi, 0, 1, 2, 3, 4, 5, 6, 7,
                                           8, 9, 10, 11, 12, 13, 14, 15);
}

__device__ __forceinline__ v8f wmma_f16(v16h a, v16h b, v8f c) {
    return __builtin_amdgcn_wmma_f32_16x16x32_f16(false, a, false, b,
                                                  (short)0, c, false, false);
}

__device__ __forceinline__ float sigmoidf_fast(float x) {
    return 1.0f / (1.0f + __expf(-x));
}

// ---- 1. f32 -> f16 conversion ----------------------------------------------
__global__ void cvt_f32_f16_kernel(const float* __restrict__ src,
                                   _Float16* __restrict__ dst, long n) {
    long i = (long)blockIdx.x * blockDim.x + threadIdx.x;
    long stride = (long)gridDim.x * blockDim.x;
    for (; i < n; i += stride) dst[i] = (_Float16)src[i];
}

// ---- 2. gx = X(M,K) @ W(N,K)^T + bih + bhh  (f32 out) ----------------------
// One 16x64 output strip per wave (4 accs, A fragment reused 4x), 8 waves/block.
__global__ __launch_bounds__(256)
void gemm_xw_bias_kernel(const _Float16* __restrict__ X,
                         const _Float16* __restrict__ W,
                         const float* __restrict__ bih,
                         const float* __restrict__ bhh,
                         float* __restrict__ out, int M, int N, int K) {
    const int lane = threadIdx.x & 31;
    const int tid  = blockIdx.x * (blockDim.x >> 5) + (threadIdx.x >> 5);
    const int ng   = N >> 6;                   // groups of 4 n-tiles
    if (tid >= (M >> 4) * ng) return;
    const int tm = tid / ng, tg = tid % ng;
    const _Float16* Xb = X + (long)tm * 16 * K;
    const _Float16* Wb = W + (long)tg * 64 * K;
    v8f acc[4] = {{}, {}, {}, {}};
    for (int kb = 0; kb < K; kb += 32) {
        v16h a  = load_a16(Xb, K, kb, lane);
        v16h b0 = load_b16(Wb + 0L * 16 * K, K, kb, lane);
        v16h b1 = load_b16(Wb + 1L * 16 * K, K, kb, lane);
        v16h b2 = load_b16(Wb + 2L * 16 * K, K, kb, lane);
        v16h b3 = load_b16(Wb + 3L * 16 * K, K, kb, lane);
        acc[0] = wmma_f16(a, b0, acc[0]);
        acc[1] = wmma_f16(a, b1, acc[1]);
        acc[2] = wmma_f16(a, b2, acc[2]);
        acc[3] = wmma_f16(a, b3, acc[3]);
    }
#pragma unroll
    for (int q = 0; q < 4; ++q) {
        const int n = tg * 64 + q * 16 + (lane & 15);
        const float bias = bih[n] + bhh[n];
#pragma unroll
        for (int r = 0; r < 8; ++r) {
            const int m = tm * 16 + r + ((lane >> 4) << 3);
            out[(long)m * N + n] = acc[q][r] + bias;
        }
    }
}

// ---- 3. LSTM recurrence -----------------------------------------------------
// grid = 4 blocks (16 batch rows each), block = 512 threads = 16 waves.
// Wave jt owns hidden columns [jt*16, jt*16+16) and computes the FOUR gate
// tiles for those columns, so gate math + cell state c stay in registers in
// accumulator layout. Only h (16x256 f16 = 8KB) goes through LDS per step.
__global__ __launch_bounds__(512)
void lstm_rec_kernel(const float* __restrict__ gx,       // (steps, B, 4H)
                     const _Float16* __restrict__ Whh,   // (4H, H) row-major
                     const float* __restrict__ h0,       // (B, H) or null
                     const float* __restrict__ c0,       // (B, H) or null
                     _Float16* __restrict__ out16,       // (steps, B, H)
                     float* __restrict__ outf32,         // opt, row stride ostride
                     int ostride,
                     float* __restrict__ hN,             // opt (B, H)
                     float* __restrict__ cN,             // opt (B, H)
                     int steps) {
    __shared__ _Float16 h_sh[16 * HID];
    const int lane  = threadIdx.x & 31;
    const int jt    = threadIdx.x >> 5;     // hidden tile 0..15
    const int bbase = blockIdx.x * 16;
    const int j     = jt * 16 + (lane & 15);

    for (int idx = threadIdx.x; idx < 16 * HID; idx += blockDim.x) {
        const int r = idx / HID, cc = idx % HID;
        const float v = h0 ? h0[(long)(bbase + r) * HID + cc] : 0.0f;
        h_sh[idx] = (_Float16)v;
    }
    v8f creg, hreg;
#pragma unroll
    for (int r = 0; r < 8; ++r) {
        const int row = r + ((lane >> 4) << 3);
        creg[r] = c0 ? c0[(long)(bbase + row) * HID + j] : 0.0f;
        hreg[r] = h0 ? h0[(long)(bbase + row) * HID + j] : 0.0f;
    }
    __syncthreads();

    const _Float16* Wg0 = Whh + (long)(0 * HID + jt * 16) * HID;
    const _Float16* Wg1 = Whh + (long)(1 * HID + jt * 16) * HID;
    const _Float16* Wg2 = Whh + (long)(2 * HID + jt * 16) * HID;
    const _Float16* Wg3 = Whh + (long)(3 * HID + jt * 16) * HID;

    for (int s = 0; s < steps; ++s) {
        const float* gxs = gx + ((long)s * BATCH + bbase) * G4H;
        if (s + 1 < steps)  // pull next step's gate pre-activations toward L2/L0
            __builtin_prefetch(gx + ((long)(s + 1) * BATCH + bbase) * G4H, 0, 3);
        v8f acc[4];
#pragma unroll
        for (int g = 0; g < 4; ++g)
#pragma unroll
            for (int r = 0; r < 8; ++r) {
                const int row = r + ((lane >> 4) << 3);
                acc[g][r] = gxs[(long)row * G4H + g * HID + j];
            }
#pragma unroll
        for (int kb = 0; kb < HID; kb += 32) {
            v16h a  = load_a16(h_sh, HID, kb, lane);      // LDS ds_load
            v16h b0 = load_b16(Wg0, HID, kb, lane);
            v16h b1 = load_b16(Wg1, HID, kb, lane);
            v16h b2 = load_b16(Wg2, HID, kb, lane);
            v16h b3 = load_b16(Wg3, HID, kb, lane);
            acc[0] = wmma_f16(a, b0, acc[0]);
            acc[1] = wmma_f16(a, b1, acc[1]);
            acc[2] = wmma_f16(a, b2, acc[2]);
            acc[3] = wmma_f16(a, b3, acc[3]);
        }
        __syncthreads();   // all reads of h_sh (h_{s-1}) complete
#pragma unroll
        for (int r = 0; r < 8; ++r) {
            const int row = r + ((lane >> 4) << 3);
            const float ig = sigmoidf_fast(acc[0][r]);
            const float fg = sigmoidf_fast(acc[1][r]);
            const float gg = tanhf(acc[2][r]);
            const float og = sigmoidf_fast(acc[3][r]);
            const float c  = fg * creg[r] + ig * gg;
            creg[r] = c;
            const float h = og * tanhf(c);
            hreg[r] = h;
            h_sh[row * HID + j] = (_Float16)h;
            const long orow = (long)s * BATCH + bbase + row;
            out16[orow * HID + j] = (_Float16)h;
            if (outf32) outf32[orow * (long)ostride + j] = h;
        }
        __syncthreads();   // h_s visible to all waves
    }
#pragma unroll
    for (int r = 0; r < 8; ++r) {
        const int row = r + ((lane >> 4) << 3);
        if (hN) hN[(long)(bbase + row) * HID + j] = hreg[r];
        if (cN) cN[(long)(bbase + row) * HID + j] = creg[r];
    }
}

// ---- 4. enc_out transpose (S,B,H) -> (B,H,S) --------------------------------
// 64x64 (s x h) tile per block, staged through LDS with CDNA5 async copy.
// `tile` is the only LDS allocation -> wave LDS offset 0 addressing.
__global__ __launch_bounds__(256)
void transpose_bhs_kernel(const _Float16* __restrict__ in,   // (S,B,H)
                          _Float16* __restrict__ out) {      // (B,H,S)
    __shared__ _Float16 tile[64 * 72];      // pad 64->72 (16B-aligned rows)
    const int tS = S_LEN / 64, tH = HID / 64;            // 8, 4
    const int b   = blockIdx.x / (tS * tH);
    const int rem = blockIdx.x % (tS * tH);
    const int s0 = (rem / tH) * 64;
    const int h0 = (rem % tH) * 64;
#pragma unroll
    for (int k = 0; k < 2; ++k) {           // 512 x 16B chunks, 256 threads
        const int ci = threadIdx.x + k * 256;
        const int r  = ci >> 2;             // s row within tile
        const int c  = (ci & 3) << 4;       // h offset within tile (16 halves)
        const _Float16* gp = in + ((long)(s0 + r) * BATCH + b) * HID + h0 + c;
        const unsigned loff = (unsigned)((r * 72 + c) * 2);
        asm volatile("global_load_async_to_lds_b128 %0, %1, off"
                     :: "v"(loff), "v"(gp) : "memory");
    }
    asm volatile("s_wait_asynccnt 0x0" ::: "memory");
    __syncthreads();
    const int h  = threadIdx.x >> 2;        // 0..63
    const int sc = (threadIdx.x & 3) << 4;  // 0,16,32,48
    v16h v;
#pragma unroll
    for (int i = 0; i < 16; ++i) v[i] = tile[(sc + i) * 72 + h];
    *(v16h*)(out + ((long)b * HID + h0 + h) * S_LEN + s0 + sc) = v;
}

// ---- 5a. scores[t,b,s] = sum_h dec[t,b,h] * enc[s,b,h] ---------------------
// 16x64 strip per wave (4 s-tiles share one A fragment).
__global__ __launch_bounds__(256)
void attn_scores_kernel(const _Float16* __restrict__ dec16,  // (T,B,H)
                        const _Float16* __restrict__ enc16,  // (S,B,H)
                        float* __restrict__ scores) {        // (T,B,S)
    const int lane = threadIdx.x & 31;
    const int tid  = blockIdx.x * 8 + (threadIdx.x >> 5);
    const int TT = T_LEN / 16, SG = S_LEN / 64;              // 32, 8
    if (tid >= BATCH * TT * SG) return;
    const int b   = tid / (TT * SG);
    const int rem = tid % (TT * SG);
    const int tt = rem / SG, tsg = rem % SG;
    const _Float16* Ab = dec16 + ((long)tt * 16 * BATCH + b) * HID;
    const _Float16* B0 = enc16 + ((long)(tsg * 4 + 0) * 16 * BATCH + b) * HID;
    const _Float16* B1 = enc16 + ((long)(tsg * 4 + 1) * 16 * BATCH + b) * HID;
    const _Float16* B2 = enc16 + ((long)(tsg * 4 + 2) * 16 * BATCH + b) * HID;
    const _Float16* B3 = enc16 + ((long)(tsg * 4 + 3) * 16 * BATCH + b) * HID;
    v8f acc[4] = {{}, {}, {}, {}};
    for (int kb = 0; kb < HID; kb += 32) {
        v16h a  = load_a16(Ab, (long)BATCH * HID, kb, lane);
        v16h b0 = load_b16(B0, (long)BATCH * HID, kb, lane);
        v16h b1 = load_b16(B1, (long)BATCH * HID, kb, lane);
        v16h b2 = load_b16(B2, (long)BATCH * HID, kb, lane);
        v16h b3 = load_b16(B3, (long)BATCH * HID, kb, lane);
        acc[0] = wmma_f16(a, b0, acc[0]);
        acc[1] = wmma_f16(a, b1, acc[1]);
        acc[2] = wmma_f16(a, b2, acc[2]);
        acc[3] = wmma_f16(a, b3, acc[3]);
    }
#pragma unroll
    for (int q = 0; q < 4; ++q) {
        const int scol = (tsg * 4 + q) * 16 + (lane & 15);
#pragma unroll
        for (int r = 0; r < 8; ++r) {
            const int trow = tt * 16 + r + ((lane >> 4) << 3);
            scores[((long)trow * BATCH + b) * S_LEN + scol] = acc[q][r];
        }
    }
}

// ---- 5b. softmax over s, writes f32 (in place, = d_out attn region) + f16 --
__global__ __launch_bounds__(128)
void softmax_kernel(float* __restrict__ scores, _Float16* __restrict__ w16) {
    const long row = blockIdx.x;              // (t*B + b)
    float* p = scores + row * S_LEN;
    __shared__ float red[128];
    const int tid = threadIdx.x;
    float vals[4];
    float mx = -3.0e38f;
#pragma unroll
    for (int i = 0; i < 4; ++i) { vals[i] = p[tid + i * 128]; mx = fmaxf(mx, vals[i]); }
    red[tid] = mx; __syncthreads();
    for (int off = 64; off > 0; off >>= 1) {
        if (tid < off) red[tid] = fmaxf(red[tid], red[tid + off]);
        __syncthreads();
    }
    mx = red[0]; __syncthreads();
    float sum = 0.0f;
#pragma unroll
    for (int i = 0; i < 4; ++i) { vals[i] = __expf(vals[i] - mx); sum += vals[i]; }
    red[tid] = sum; __syncthreads();
    for (int off = 64; off > 0; off >>= 1) {
        if (tid < off) red[tid] += red[tid + off];
        __syncthreads();
    }
    const float inv = 1.0f / red[0];
#pragma unroll
    for (int i = 0; i < 4; ++i) {
        const float w = vals[i] * inv;
        p[tid + i * 128] = w;
        w16[row * S_LEN + tid + i * 128] = (_Float16)w;
    }
}

// ---- 6. attn_v[t,b,h] = sum_s w[t,b,s] * encT[b,h,s] -> d_out cols [H,2H) --
// encT is (B,H,S) so each lane's B fragment is two contiguous 16B loads. 4 h-tiles per wave.
__global__ __launch_bounds__(256)
void attn_v_kernel(const _Float16* __restrict__ w16,    // (T,B,S)
                   const _Float16* __restrict__ encT,   // (B,H,S)
                   float* __restrict__ resp) {          // (T,B,2H)
    const int lane = threadIdx.x & 31;
    const int tid  = blockIdx.x * 8 + (threadIdx.x >> 5);
    const int TT = T_LEN / 16, HG = HID / 64;            // 32, 4
    if (tid >= BATCH * TT * HG) return;
    const int b   = tid / (TT * HG);
    const int rem = tid % (TT * HG);
    const int tt = rem / HG, hg = rem % HG;
    const _Float16* Ab = w16 + ((long)tt * 16 * BATCH + b) * S_LEN;
    const _Float16* Bb = encT + ((long)b * HID + hg * 64) * S_LEN;
    v8f acc[4] = {{}, {}, {}, {}};
    for (int kb = 0; kb < S_LEN; kb += 32) {
        v16h a  = load_a16(Ab, (long)BATCH * S_LEN, kb, lane);
        v16h b0 = load_b16(Bb + 0L * 16 * S_LEN, S_LEN, kb, lane);
        v16h b1 = load_b16(Bb + 1L * 16 * S_LEN, S_LEN, kb, lane);
        v16h b2 = load_b16(Bb + 2L * 16 * S_LEN, S_LEN, kb, lane);
        v16h b3 = load_b16(Bb + 3L * 16 * S_LEN, S_LEN, kb, lane);
        acc[0] = wmma_f16(a, b0, acc[0]);
        acc[1] = wmma_f16(a, b1, acc[1]);
        acc[2] = wmma_f16(a, b2, acc[2]);
        acc[3] = wmma_f16(a, b3, acc[3]);
    }
#pragma unroll
    for (int q = 0; q < 4; ++q) {
        const int n = (hg * 4 + q) * 16 + (lane & 15);
#pragma unroll
        for (int r = 0; r < 8; ++r) {
            const int trow = tt * 16 + r + ((lane >> 4) << 3);
            resp[((long)trow * BATCH + b) * (2 * HID) + HID + n] = acc[q][r];
        }
    }
}

// ---------------------------------------------------------------------------
extern "C" void kernel_launch(void* const* d_in, const int* in_sizes, int n_in,
                              void* d_out, int out_size, void* d_ws, size_t ws_size,
                              hipStream_t stream) {
    (void)in_sizes; (void)n_in; (void)out_size; (void)ws_size;
    const float* enc_input = (const float*)d_in[0];
    const float* dec_input = (const float*)d_in[1];
    const float* enc_Wih   = (const float*)d_in[2];
    const float* enc_Whh   = (const float*)d_in[3];
    const float* enc_bih   = (const float*)d_in[4];
    const float* enc_bhh   = (const float*)d_in[5];
    const float* dec_Wih   = (const float*)d_in[6];
    const float* dec_Whh   = (const float*)d_in[7];
    const float* dec_bih   = (const float*)d_in[8];
    const float* dec_bhh   = (const float*)d_in[9];

    // Workspace carve-up (~257 MB total)
    char* w = (char*)d_ws;
    auto carve = [&](size_t bytes) { char* p = w; w += (bytes + 255) & ~(size_t)255; return p; };
    const long WSZ = 2L * G4H * HID;                       // one weight tensor, both layers
    _Float16* xin16    = (_Float16*)carve((long)SB * INDIM * 2);
    _Float16* l0out16  = (_Float16*)carve((long)SB * HID * 2);
    _Float16* encout16 = (_Float16*)carve((long)SB * HID * 2);
    _Float16* decout16 = (_Float16*)carve((long)SB * HID * 2);
    _Float16* encT16   = (_Float16*)carve((long)BATCH * HID * S_LEN * 2);
    _Float16* w16      = (_Float16*)carve(4 * WSZ * 2);
    _Float16* attnw16  = (_Float16*)carve((long)T_LEN * BATCH * S_LEN * 2);
    float*    gx       = (float*)   carve((long)SB * G4H * 4);
    float*    st       = (float*)   carve(4L * BATCH * HID * 4);

    _Float16* eWih = w16;            // (2, 4H, 256)
    _Float16* eWhh = w16 + WSZ;
    _Float16* dWih = w16 + 2 * WSZ;
    _Float16* dWhh = w16 + 3 * WSZ;
    const long LW = (long)G4H * HID;                       // per-layer weight elems
    float* eh0 = st;                                       // enc final h, layer 0
    float* eh1 = st + (long)BATCH * HID;
    float* ec0 = st + 2L * BATCH * HID;
    float* ec1 = st + 3L * BATCH * HID;

    const int gemmBlocks = ((SB / 16) * (G4H / 64)) / 8;   // 4096

    // ---- convert weights + encoder input to f16
    cvt_f32_f16_kernel<<<2048, 256, 0, stream>>>(enc_input, xin16, (long)SB * INDIM);
    cvt_f32_f16_kernel<<<1024, 256, 0, stream>>>(enc_Wih, eWih, WSZ);
    cvt_f32_f16_kernel<<<1024, 256, 0, stream>>>(enc_Whh, eWhh, WSZ);
    cvt_f32_f16_kernel<<<1024, 256, 0, stream>>>(dec_Wih, dWih, WSZ);
    cvt_f32_f16_kernel<<<1024, 256, 0, stream>>>(dec_Whh, dWhh, WSZ);

    // ---- encoder layer 0
    gemm_xw_bias_kernel<<<gemmBlocks, 256, 0, stream>>>(xin16, eWih, enc_bih, enc_bhh,
                                                        gx, SB, G4H, INDIM);
    lstm_rec_kernel<<<4, 512, 0, stream>>>(gx, eWhh, nullptr, nullptr, l0out16,
                                           nullptr, 0, eh0, ec0, S_LEN);
    // ---- encoder layer 1
    gemm_xw_bias_kernel<<<gemmBlocks, 256, 0, stream>>>(l0out16, eWih + LW,
                                                        enc_bih + G4H, enc_bhh + G4H,
                                                        gx, SB, G4H, HID);
    lstm_rec_kernel<<<4, 512, 0, stream>>>(gx, eWhh + LW, nullptr, nullptr, encout16,
                                           nullptr, 0, eh1, ec1, S_LEN);
    // ---- decoder layer 0 (init from encoder final state)
    cvt_f32_f16_kernel<<<2048, 256, 0, stream>>>(dec_input, xin16, (long)SB * INDIM);
    gemm_xw_bias_kernel<<<gemmBlocks, 256, 0, stream>>>(xin16, dWih, dec_bih, dec_bhh,
                                                        gx, SB, G4H, INDIM);
    lstm_rec_kernel<<<4, 512, 0, stream>>>(gx, dWhh, eh0, ec0, l0out16,
                                           nullptr, 0, nullptr, nullptr, T_LEN);
    // ---- decoder layer 1: h goes straight into d_out responses cols [0,H)
    gemm_xw_bias_kernel<<<gemmBlocks, 256, 0, stream>>>(l0out16, dWih + LW,
                                                        dec_bih + G4H, dec_bhh + G4H,
                                                        gx, SB, G4H, HID);
    lstm_rec_kernel<<<4, 512, 0, stream>>>(gx, dWhh + LW, eh1, ec1, decout16,
                                           (float*)d_out, 2 * HID, nullptr, nullptr, T_LEN);

    // ---- attention
    transpose_bhs_kernel<<<BATCH * (S_LEN / 64) * (HID / 64), 256, 0, stream>>>(
        encout16, encT16);
    float* attn_region = (float*)d_out + (long)T_LEN * BATCH * 2 * HID;
    attn_scores_kernel<<<(BATCH * 32 * 8) / 8, 256, 0, stream>>>(decout16, encout16,
                                                                 attn_region);
    softmax_kernel<<<T_LEN * BATCH, 128, 0, stream>>>(attn_region, attnw16);
    attn_v_kernel<<<(BATCH * 32 * 4) / 8, 256, 0, stream>>>(attnw16, encT16,
                                                            (float*)d_out);
}
